// SceneGraphViT_4913442586857
// MI455X (gfx1250) — compile-verified, matrix operand order checked by reading, not run on hardware
//
#include <hip/hip_runtime.h>
#include <hip/hip_bf16.h>
#include <math.h>

// ---------------------------------------------------------------------------
// Problem constants (from reference)
// ---------------------------------------------------------------------------
#define BATCH      2
#define NTOK       1024
#define DIM        768
#define KINST      512
#define NCLS       151      // NUM_CLASSES + 1
#define LN_EPS     1e-5f

typedef __attribute__((ext_vector_type(16))) _Float16 v16h;
typedef __attribute__((ext_vector_type(8)))  _Float16 v8h;
typedef __attribute__((ext_vector_type(8)))  float    v8f;

union AFrag { v16h v; v8h h[2]; };

#define ACT_NONE 0
#define ACT_GELU 1
#define ACT_RELU 2

__device__ inline float gelu_exact(float v) {
    return 0.5f * v * (1.0f + erff(v * 0.70710678118654752440f));
}

__device__ inline v16h load_frag(const _Float16* p) {
    AFrag u;
    u.h[0] = *(const v8h*)(p);
    u.h[1] = *(const v8h*)(p + 16);
    return u.v;
}

// ---------------------------------------------------------------------------
// f32 -> f16 conversion with zero padding: dst[i] = (i < n) ? src[i] : 0
// for i < npad. Padding lets the GEMM drop all bounds checks on B loads.
// ---------------------------------------------------------------------------
__global__ __launch_bounds__(256) void cvt_f32_to_f16_kernel(
    const float* __restrict__ src, _Float16* __restrict__ dst, int n, int npad)
{
    int i = blockIdx.x * 256 + threadIdx.x;
    if (i < npad) dst[i] = (i < n) ? (_Float16)src[i] : (_Float16)0.0f;
}

// ---------------------------------------------------------------------------
// WMMA GEMM:  C[M,N] = act( A[M,K] @ W[N,K]^T + bias )
//   A: f16 row-major (lda = K), W: f16 row-major (ldw = K)  -> B = W^T
// Block = 256 threads = 8 waves; block tile 128 x 128; wave tile 32 x 64
// (2x4 fragments of v_wmma_f32_16x16x32_f16 -> 8 WMMA / 12 b128 loads per
// K-step). M must be a multiple of 128. W must be zero-padded up to
// gridDim.y*128 rows so B loads need no bounds checks; stores guarded by N.
// Epilogue fully specialized at compile time (no runtime branches).
// ---------------------------------------------------------------------------
template <int ACT, bool HAS_BIAS, bool OUT_H, bool OUT_F>
__device__ inline void store_frag(v8f acc, int row0, int col, int N,
                                  const float* __restrict__ bias,
                                  _Float16* __restrict__ outH,
                                  float* __restrict__ outF)
{
    if (col >= N) return;
    float bv = HAS_BIAS ? bias[col] : 0.0f;
#pragma unroll
    for (int r = 0; r < 8; ++r) {
        float v = acc[r] + bv;
        if (ACT == ACT_GELU)      v = gelu_exact(v);
        else if (ACT == ACT_RELU) v = fmaxf(v, 0.0f);
        size_t idx = (size_t)(row0 + r) * N + col;
        if (OUT_H) outH[idx] = (_Float16)v;
        if (OUT_F) outF[idx] = v;
    }
}

template <int ACT, bool HAS_BIAS, bool OUT_H, bool OUT_F>
__global__ __launch_bounds__(256) void wmma_gemm_kernel(
    const _Float16* __restrict__ A, const _Float16* __restrict__ W,
    const float* __restrict__ bias,
    _Float16* __restrict__ outH, float* __restrict__ outF,
    int M, int N, int K)
{
    const int wave = threadIdx.x >> 5;      // 0..7
    const int lane = threadIdx.x & 31;
    const int wm   = wave & 3;              // 4 waves along M (32 rows each)
    const int wn   = wave >> 2;             // 2 waves along N (64 cols each)
    const int rowBase = blockIdx.x * 128 + wm * 32;
    const int colBase = blockIdx.y * 128 + wn * 64;
    const int m  = lane & 15;               // row (A) / col (B) within 16-tile
    const int kh = lane >> 4;               // K-half selector

    // Named scalar offsets (no pointer arrays -> keeps global addrspace
    // provenance on the kernarg base pointers -> global_load_b128, not flat).
    const size_t a0 = (size_t)(rowBase + m) * K + kh * 8;
    const size_t a1 = a0 + (size_t)16 * K;
    const size_t b0 = (size_t)(colBase + m) * K + kh * 8;
    const size_t b1 = b0 + (size_t)16 * K;
    const size_t b2 = b0 + (size_t)32 * K;
    const size_t b3 = b0 + (size_t)48 * K;

    v8f acc[2][4];
#pragma unroll
    for (int ti = 0; ti < 2; ++ti)
#pragma unroll
        for (int tj = 0; tj < 4; ++tj)
#pragma unroll
            for (int e = 0; e < 8; ++e) acc[ti][tj][e] = 0.0f;

    for (int k0 = 0; k0 < K; k0 += 32) {
        v16h fa0 = load_frag(A + a0 + k0);
        v16h fa1 = load_frag(A + a1 + k0);
        v16h fb0 = load_frag(W + b0 + k0);
        v16h fb1 = load_frag(W + b1 + k0);
        v16h fb2 = load_frag(W + b2 + k0);
        v16h fb3 = load_frag(W + b3 + k0);

        acc[0][0] = __builtin_amdgcn_wmma_f32_16x16x32_f16(false, fa0, false, fb0, (short)0, acc[0][0], false, false);
        acc[0][1] = __builtin_amdgcn_wmma_f32_16x16x32_f16(false, fa0, false, fb1, (short)0, acc[0][1], false, false);
        acc[0][2] = __builtin_amdgcn_wmma_f32_16x16x32_f16(false, fa0, false, fb2, (short)0, acc[0][2], false, false);
        acc[0][3] = __builtin_amdgcn_wmma_f32_16x16x32_f16(false, fa0, false, fb3, (short)0, acc[0][3], false, false);
        acc[1][0] = __builtin_amdgcn_wmma_f32_16x16x32_f16(false, fa1, false, fb0, (short)0, acc[1][0], false, false);
        acc[1][1] = __builtin_amdgcn_wmma_f32_16x16x32_f16(false, fa1, false, fb1, (short)0, acc[1][1], false, false);
        acc[1][2] = __builtin_amdgcn_wmma_f32_16x16x32_f16(false, fa1, false, fb2, (short)0, acc[1][2], false, false);
        acc[1][3] = __builtin_amdgcn_wmma_f32_16x16x32_f16(false, fa1, false, fb3, (short)0, acc[1][3], false, false);
    }

    // Epilogue: C layout -> row = 16*ti + 8*kh + r, col = 16*tj + m
#pragma unroll
    for (int ti = 0; ti < 2; ++ti) {
        const int row0 = rowBase + 16 * ti + 8 * kh;
#pragma unroll
        for (int tj = 0; tj < 4; ++tj) {
            const int col = colBase + 16 * tj + m;
            store_frag<ACT, HAS_BIAS, OUT_H, OUT_F>(acc[ti][tj], row0, col, N,
                                                    bias, outH, outF);
        }
    }
}

// ---------------------------------------------------------------------------
// LayerNorm (+ optional residual):  out = xres + (LN(h)*g + be)
// one 256-thread block per row of length d
// ---------------------------------------------------------------------------
__global__ __launch_bounds__(256) void ln_kernel(
    const float* __restrict__ h, const float* __restrict__ xres,
    const float* __restrict__ g, const float* __restrict__ be,
    float* __restrict__ outF, _Float16* __restrict__ outH, int d)
{
    const size_t row = blockIdx.x;
    h += row * d;
    if (xres) xres += row * d;
    const int tid = threadIdx.x;
    __shared__ float redA[256];
    __shared__ float redB[256];

    float s = 0.0f, ss = 0.0f;
    for (int j = tid; j < d; j += 256) { float v = h[j]; s += v; ss += v * v; }
    redA[tid] = s; redB[tid] = ss;
    __syncthreads();
    for (int off = 128; off > 0; off >>= 1) {
        if (tid < off) { redA[tid] += redA[tid + off]; redB[tid] += redB[tid + off]; }
        __syncthreads();
    }
    const float mu  = redA[0] / (float)d;
    const float var = redB[0] / (float)d - mu * mu;
    const float inv = rsqrtf(var + LN_EPS);

    for (int j = tid; j < d; j += 256) {
        float y = (h[j] - mu) * inv * g[j] + be[j];
        float o = xres ? (xres[j] + y) : y;
        if (outF) outF[row * d + j] = o;
        if (outH) outH[row * d + j] = (_Float16)o;
    }
}

// ---------------------------------------------------------------------------
// Per-row softmax over full score row, keep only the diagonal probability.
// scores layout [b*NTOK + i][NTOK]; one block per row.
// ---------------------------------------------------------------------------
__global__ __launch_bounds__(256) void softmax_diag_kernel(
    const float* __restrict__ scores, float* __restrict__ diag, int n)
{
    const int row = blockIdx.x;             // b*n + i
    const int i   = row % n;
    const float* s = scores + (size_t)row * n;
    const int tid = threadIdx.x;
    __shared__ float red[256];

    float mx = -3.4e38f;
    for (int j = tid; j < n; j += 256) mx = fmaxf(mx, s[j]);
    red[tid] = mx; __syncthreads();
    for (int off = 128; off > 0; off >>= 1) {
        if (tid < off) red[tid] = fmaxf(red[tid], red[tid + off]);
        __syncthreads();
    }
    mx = red[0]; __syncthreads();

    float e = 0.0f;
    for (int j = tid; j < n; j += 256) e += expf(s[j] - mx);
    red[tid] = e; __syncthreads();
    for (int off = 128; off > 0; off >>= 1) {
        if (tid < off) red[tid] += red[tid + off];
        __syncthreads();
    }
    if (tid == 0) diag[row] = expf(s[i] - mx) / red[0];
}

// ---------------------------------------------------------------------------
// Top-K by rank counting (exact, deterministic; tie-break = lower index wins,
// matching jax.lax.top_k). flags[b*n+i] = 1 if i is in top-kinst of diag row.
// ---------------------------------------------------------------------------
__global__ __launch_bounds__(256) void topk_flag_kernel(
    const float* __restrict__ diag, int* __restrict__ flags, int n, int kinst)
{
    int t = blockIdx.x * 256 + threadIdx.x;
    if (t >= BATCH * n) return;
    int b = t / n, i = t % n;
    const float* d = diag + (size_t)b * n;
    float v = d[i];
    int cnt = 0;
    for (int j = 0; j < n; ++j) {
        float dj = d[j];
        cnt += (dj > v) || (dj == v && j < i);
    }
    flags[t] = (cnt < kinst) ? 1 : 0;
}

// Selected indices, ascending (matches jnp.sort(tk)): position = #flags before i
__global__ __launch_bounds__(256) void topk_scatter_kernel(
    const int* __restrict__ flags, int* __restrict__ tk, int n, int kinst)
{
    int t = blockIdx.x * 256 + threadIdx.x;
    if (t >= BATCH * n) return;
    if (!flags[t]) return;
    int b = t / n, i = t % n;
    const int* f = flags + (size_t)b * n;
    int pos = 0;
    for (int j = 0; j < i; ++j) pos += f[j];
    tk[b * kinst + pos] = i;
}

// ---------------------------------------------------------------------------
// Build mlp2 input for the self pairs: rel_in = LN(2 * q[b, tk[idx], :])
// (subj_e + obj_e with subj_tok == obj_tok; LN has no affine here)
// one block per selected instance (B*KINST rows)
// ---------------------------------------------------------------------------
__global__ __launch_bounds__(256) void build_rel_kernel(
    const float* __restrict__ q, const int* __restrict__ tk,
    _Float16* __restrict__ rel_h, int n, int d, int kinst)
{
    const int row = blockIdx.x;             // b*kinst + idx
    const int b = row / kinst;
    const int t = tk[row];
    const float* e = q + ((size_t)b * n + t) * d;
    const int tid = threadIdx.x;
    __shared__ float redA[256];
    __shared__ float redB[256];

    float s = 0.0f, ss = 0.0f;
    for (int j = tid; j < d; j += 256) {
        float v = 2.0f * e[j];
        s += v; ss += v * v;
    }
    redA[tid] = s; redB[tid] = ss;
    __syncthreads();
    for (int off = 128; off > 0; off >>= 1) {
        if (tid < off) { redA[tid] += redA[tid + off]; redB[tid] += redB[tid + off]; }
        __syncthreads();
    }
    const float mu  = redA[0] / (float)d;
    const float var = redB[0] / (float)d - mu * mu;
    const float inv = rsqrtf(var + LN_EPS);

    for (int j = tid; j < d; j += 256) {
        float v = 2.0f * e[j];
        rel_h[(size_t)row * d + j] = (_Float16)((v - mu) * inv);
    }
}

// ---------------------------------------------------------------------------
// Row softmax for class logits (ncls = 151); one block per row.
// ---------------------------------------------------------------------------
__global__ __launch_bounds__(256) void softmax_rows_kernel(
    const float* __restrict__ logits, float* __restrict__ out, int ncls)
{
    const int row = blockIdx.x;
    const float* l = logits + (size_t)row * ncls;
    const int tid = threadIdx.x;
    __shared__ float red[256];

    float mx = -3.4e38f;
    for (int j = tid; j < ncls; j += 256) mx = fmaxf(mx, l[j]);
    red[tid] = mx; __syncthreads();
    for (int off = 128; off > 0; off >>= 1) {
        if (tid < off) red[tid] = fmaxf(red[tid], red[tid + off]);
        __syncthreads();
    }
    mx = red[0]; __syncthreads();

    float e = 0.0f;
    for (int j = tid; j < ncls; j += 256) e += expf(l[j] - mx);
    red[tid] = e; __syncthreads();
    for (int off = 128; off > 0; off >>= 1) {
        if (tid < off) red[tid] += red[tid + off];
        __syncthreads();
    }
    const float inv = 1.0f / red[0];
    for (int j = tid; j < ncls; j += 256)
        out[(size_t)row * ncls + j] = expf(l[j] - mx) * inv;
}

// ---------------------------------------------------------------------------
// Host side
// ---------------------------------------------------------------------------
static inline size_t align_up(size_t x) { return (x + 255) & ~(size_t)255; }

extern "C" void kernel_launch(void* const* d_in, const int* in_sizes, int n_in,
                              void* d_out, int out_size, void* d_ws, size_t ws_size,
                              hipStream_t stream) {
    (void)in_sizes; (void)n_in; (void)out_size; (void)ws_size;

    // Input order: 'x' first, then params flattened in JAX pytree (alphabetical
    // dict-key) order:
    //   bbox_b, bbox_w, cls_b, cls_w,
    //   mlp2{b1,b2,be,g,w1,w2},
    //   object_head{b1,b2,b3,be,g,w1,w2,w3},
    //   subject_head{b1,b2,b3,be,g,w1,w2,w3}
    const float* x      = (const float*)d_in[0];
    const float* bbox_b = (const float*)d_in[1];
    const float* bbox_w = (const float*)d_in[2];
    const float* cls_b  = (const float*)d_in[3];
    const float* cls_w  = (const float*)d_in[4];
    const float* m2_b1  = (const float*)d_in[5];
    const float* m2_b2  = (const float*)d_in[6];
    const float* m2_be  = (const float*)d_in[7];
    const float* m2_g   = (const float*)d_in[8];
    const float* m2_w1  = (const float*)d_in[9];
    const float* m2_w2  = (const float*)d_in[10];
    const float* ob_b1  = (const float*)d_in[11];
    const float* ob_b2  = (const float*)d_in[12];
    const float* ob_b3  = (const float*)d_in[13];
    const float* ob_be  = (const float*)d_in[14];
    const float* ob_g   = (const float*)d_in[15];
    const float* ob_w1  = (const float*)d_in[16];
    const float* ob_w2  = (const float*)d_in[17];
    const float* ob_w3  = (const float*)d_in[18];
    const float* sb_b1  = (const float*)d_in[19];
    const float* sb_b2  = (const float*)d_in[20];
    const float* sb_b3  = (const float*)d_in[21];
    const float* sb_be  = (const float*)d_in[22];
    const float* sb_g   = (const float*)d_in[23];
    const float* sb_w1  = (const float*)d_in[24];
    const float* sb_w2  = (const float*)d_in[25];
    const float* sb_w3  = (const float*)d_in[26];

    const int ROWS     = BATCH * NTOK;   // 2048
    const int RROWS    = BATCH * KINST;  // 1024
    const int WSZ      = DIM * DIM;      // 589824
    const int NCLS_PAD = 256;            // cls weight rows padded to tile mult
    const int NBOX_PAD = 128;            // bbox weight rows padded to tile mult

    // Workspace layout
    char* ws = (char*)d_ws;
    size_t off = 0;
    auto carve = [&](size_t bytes) { char* p = ws + off; off = align_up(off + bytes); return p; };

    _Float16* xh     = (_Float16*)carve((size_t)ROWS * DIM * 2);
    _Float16* sw1h   = (_Float16*)carve((size_t)WSZ * 2);
    _Float16* sw2h   = (_Float16*)carve((size_t)WSZ * 2);
    _Float16* sw3h   = (_Float16*)carve((size_t)WSZ * 2);
    _Float16* ow1h   = (_Float16*)carve((size_t)WSZ * 2);
    _Float16* ow2h   = (_Float16*)carve((size_t)WSZ * 2);
    _Float16* ow3h   = (_Float16*)carve((size_t)WSZ * 2);
    _Float16* m2w1h  = (_Float16*)carve((size_t)WSZ * 2);
    _Float16* m2w2h  = (_Float16*)carve((size_t)WSZ * 2);
    _Float16* clswh  = (_Float16*)carve((size_t)NCLS_PAD * DIM * 2);
    _Float16* bboxwh = (_Float16*)carve((size_t)NBOX_PAD * DIM * 2);
    _Float16* h1h    = (_Float16*)carve((size_t)ROWS * DIM * 2);
    _Float16* h2h    = (_Float16*)carve((size_t)ROWS * DIM * 2);
    float*    h3f    = (float*)   carve((size_t)ROWS * DIM * 4);
    float*    qf     = (float*)   carve((size_t)ROWS * DIM * 4);
    _Float16* qh     = (_Float16*)carve((size_t)ROWS * DIM * 2);
    _Float16* kh     = (_Float16*)carve((size_t)ROWS * DIM * 2);
    float*    scores = (float*)   carve((size_t)BATCH * NTOK * NTOK * 4);
    float*    diag   = (float*)   carve((size_t)ROWS * 4);
    int*      flags  = (int*)     carve((size_t)ROWS * 4);
    int*      tk     = (int*)     carve((size_t)RROWS * 4);

    // Reused regions (lifetimes disjoint):
    _Float16* rel_in  = h1h;                       // B*512 x 768 f16
    _Float16* relh1   = h2h;                       // B*512 x 768 f16
    float*    relh2f  = h3f;                       // B*512 x 768 f32
    _Float16* objrelh = xh;                        // B*512 x 768 f16 (xh dead)
    float*    logits  = scores;                    // B*512 x 151 f32 (scores dead)

    float* probs    = (float*)d_out;               // (B,512,151)
    float* bbox_out = probs + (size_t)RROWS * NCLS; // (B,512,4)

    auto cvt = [&](const float* s, _Float16* d, int n, int npad) {
        cvt_f32_to_f16_kernel<<<(npad + 255) / 256, 256, 0, stream>>>(s, d, n, npad);
    };
    auto grid_for = [](int M, int N) { return dim3(M / 128, (N + 127) / 128); };

    // Specialized GEMM launchers (compile-time epilogues)
    auto gemm_gelu_h = [&](const _Float16* A, const _Float16* W, const float* bias,
                           _Float16* oH, int M, int N, int K) {
        wmma_gemm_kernel<ACT_GELU, true, true, false>
            <<<grid_for(M, N), 256, 0, stream>>>(A, W, bias, oH, nullptr, M, N, K);
    };
    auto gemm_gelu_f = [&](const _Float16* A, const _Float16* W, const float* bias,
                           float* oF, int M, int N, int K) {
        wmma_gemm_kernel<ACT_GELU, true, false, true>
            <<<grid_for(M, N), 256, 0, stream>>>(A, W, bias, nullptr, oF, M, N, K);
    };
    auto gemm_none_f = [&](const _Float16* A, const _Float16* W, const float* bias,
                           float* oF, int M, int N, int K) {
        wmma_gemm_kernel<ACT_NONE, true, false, true>
            <<<grid_for(M, N), 256, 0, stream>>>(A, W, bias, nullptr, oF, M, N, K);
    };
    auto gemm_nobias_f = [&](const _Float16* A, const _Float16* W,
                             float* oF, int M, int N, int K) {
        wmma_gemm_kernel<ACT_NONE, false, false, true>
            <<<grid_for(M, N), 256, 0, stream>>>(A, W, nullptr, nullptr, oF, M, N, K);
    };
    auto gemm_relu_f = [&](const _Float16* A, const _Float16* W, const float* bias,
                           float* oF, int M, int N, int K) {
        wmma_gemm_kernel<ACT_RELU, true, false, true>
            <<<grid_for(M, N), 256, 0, stream>>>(A, W, bias, nullptr, oF, M, N, K);
    };

    // ---- precision conversion (weights zero-padded to tile multiples) ----
    cvt(x, xh, ROWS * DIM, ROWS * DIM);
    cvt(sb_w1, sw1h, WSZ, WSZ);  cvt(sb_w2, sw2h, WSZ, WSZ);  cvt(sb_w3, sw3h, WSZ, WSZ);
    cvt(ob_w1, ow1h, WSZ, WSZ);  cvt(ob_w2, ow2h, WSZ, WSZ);  cvt(ob_w3, ow3h, WSZ, WSZ);
    cvt(m2_w1, m2w1h, WSZ, WSZ); cvt(m2_w2, m2w2h, WSZ, WSZ);
    cvt(cls_w, clswh, NCLS * DIM, NCLS_PAD * DIM);
    cvt(bbox_w, bboxwh, 4 * DIM, NBOX_PAD * DIM);

    // ---- subject head: q = x + LN(mlp3(x)) -------------------------------
    gemm_gelu_h(xh,  sw1h, sb_b1, h1h, ROWS, DIM, DIM);
    gemm_gelu_h(h1h, sw2h, sb_b2, h2h, ROWS, DIM, DIM);
    gemm_gelu_f(h2h, sw3h, sb_b3, h3f, ROWS, DIM, DIM);
    ln_kernel<<<ROWS, 256, 0, stream>>>(h3f, x, sb_g, sb_be, qf, qh, DIM);

    // ---- object head: k = x + LN(mlp3(x)) (f16 only; used for scores) ----
    gemm_gelu_h(xh,  ow1h, ob_b1, h1h, ROWS, DIM, DIM);
    gemm_gelu_h(h1h, ow2h, ob_b2, h2h, ROWS, DIM, DIM);
    gemm_gelu_f(h2h, ow3h, ob_b3, h3f, ROWS, DIM, DIM);
    ln_kernel<<<ROWS, 256, 0, stream>>>(h3f, x, ob_g, ob_be, nullptr, kh, DIM);

    // ---- scores = q @ k^T per batch; softmax diagonal --------------------
    for (int b = 0; b < BATCH; ++b) {
        gemm_nobias_f(qh + (size_t)b * NTOK * DIM, kh + (size_t)b * NTOK * DIM,
                      scores + (size_t)b * NTOK * NTOK, NTOK, NTOK, DIM);
    }
    softmax_diag_kernel<<<ROWS, 256, 0, stream>>>(scores, diag, NTOK);

    // ---- top-512 tokens per batch (rank counting, ascending order) -------
    topk_flag_kernel<<<(ROWS + 255) / 256, 256, 0, stream>>>(diag, flags, NTOK, KINST);
    topk_scatter_kernel<<<(ROWS + 255) / 256, 256, 0, stream>>>(flags, tk, NTOK, KINST);

    // ---- self-pair relation embeddings: LN(2*q[tk]) then mlp2 ------------
    build_rel_kernel<<<RROWS, 256, 0, stream>>>(qf, tk, rel_in, NTOK, DIM, KINST);
    gemm_gelu_h(rel_in, m2w1h, m2_b1, relh1, RROWS, DIM, DIM);
    gemm_none_f(relh1,  m2w2h, m2_b2, relh2f, RROWS, DIM, DIM);
    ln_kernel<<<RROWS, 256, 0, stream>>>(relh2f, nullptr, m2_g, m2_be, nullptr, objrelh, DIM);

    // ---- heads ------------------------------------------------------------
    gemm_none_f(objrelh, clswh,  cls_b,  logits,   RROWS, NCLS, DIM);
    gemm_relu_f(objrelh, bboxwh, bbox_b, bbox_out, RROWS, 4,    DIM);
    softmax_rows_kernel<<<RROWS, 256, 0, stream>>>(logits, probs, NCLS);
}
